// Linear8bitTP_43714177138985
// MI455X (gfx1250) — 8-bit matrix operand order repaired and checked by reading, compiled, not run on hardware
//
#include <hip/hip_runtime.h>
#include <hip/hip_bf16.h>
#include <stdint.h>


#define THRESH 6.0f
#define EPSQ   1e-8f

static constexpr int Bx = 4, Sx = 2048, Dk = 4096, Ox = 4096;
static constexpr int Mx = Bx * Sx;
static constexpr int WORLD = 4;
static constexpr int OUTW  = WORLD * Ox;

typedef int v8i __attribute__((ext_vector_type(8)));

static constexpr size_t WS_MASK = 0;
static constexpr size_t WS_CNT  = WS_MASK + (size_t)Dk * 4;
static constexpr size_t WS_IDX  = WS_CNT + 256;
static constexpr size_t WS_SX   = 32768;
static constexpr size_t WS_SW   = WS_SX + (size_t)Mx * 4;
static constexpr size_t WS_XQ   = 131072;
static constexpr size_t WS_WQ   = WS_XQ + (size_t)Mx * Dk;


__global__ void init_kernel(unsigned* __restrict__ mask, unsigned* __restrict__ cnt) {
    const int d = blockIdx.x * blockDim.x + threadIdx.x;
    if (d < Dk) mask[d] = 0u;
    if (d == 0) *cnt = 0u;
}

__global__ void detect_kernel(const float4* __restrict__ x4, unsigned* __restrict__ mask) {
    const size_t n4 = (size_t)Mx * Dk / 4;
    const size_t stride = (size_t)gridDim.x * blockDim.x;
    for (size_t i = (size_t)blockIdx.x * blockDim.x + threadIdx.x; i < n4; i += stride) {
        const float4 v = x4[i];
        const int d0 = (int)((i * 4) & (size_t)(Dk - 1));
        if (fabsf(v.x) > THRESH) atomicOr(&mask[d0 + 0], 1u);
        if (fabsf(v.y) > THRESH) atomicOr(&mask[d0 + 1], 1u);
        if (fabsf(v.z) > THRESH) atomicOr(&mask[d0 + 2], 1u);
        if (fabsf(v.w) > THRESH) atomicOr(&mask[d0 + 3], 1u);
    }
}

__global__ void collect_kernel(const unsigned* __restrict__ mask,
                               unsigned* __restrict__ cnt, unsigned* __restrict__ idx) {
    for (int d = threadIdx.x; d < Dk; d += blockDim.x) {
        if (mask[d]) {
            const unsigned p = atomicAdd(cnt, 1u);
            if (p < 64u) idx[p] = (unsigned)d;
        }
    }
}

__device__ __forceinline__ int quant8(float v, float inv) {
    float q = rintf(v * inv);
    q = fminf(fmaxf(q, -127.f), 127.f);
    return (int)q;
}

__global__ __launch_bounds__(256)
void quant_x_kernel(const float* __restrict__ x, const unsigned* __restrict__ mask,
                    int8_t* __restrict__ xq, float* __restrict__ sx) {
    __shared__ float red[256];
    const int m = blockIdx.x, tid = threadIdx.x;
    const float4* xr4 = (const float4*)(x + (size_t)m * Dk);
    const uint4*  mk4 = (const uint4*)mask;
    float amax = 0.f;
    for (int d4 = tid; d4 < Dk / 4; d4 += 256) {
        const float4 v = xr4[d4]; const uint4 mm = mk4[d4];
        amax = fmaxf(amax, mm.x ? 0.f : fabsf(v.x));
        amax = fmaxf(amax, mm.y ? 0.f : fabsf(v.y));
        amax = fmaxf(amax, mm.z ? 0.f : fabsf(v.z));
        amax = fmaxf(amax, mm.w ? 0.f : fabsf(v.w));
    }
    red[tid] = amax; __syncthreads();
    for (int s = 128; s > 0; s >>= 1) {
        if (tid < s) red[tid] = fmaxf(red[tid], red[tid + s]);
        __syncthreads();
    }
    const float scale = red[0] * (1.0f / 127.0f);
    if (tid == 0) sx[m] = scale;
    const float inv = 1.0f / fmaxf(scale, EPSQ);
    unsigned* qr = (unsigned*)(xq + (size_t)m * Dk);
    for (int d4 = tid; d4 < Dk / 4; d4 += 256) {
        const float4 v = xr4[d4]; const uint4 mm = mk4[d4];
        const int qx = quant8(mm.x ? 0.f : v.x, inv);
        const int qy = quant8(mm.y ? 0.f : v.y, inv);
        const int qz = quant8(mm.z ? 0.f : v.z, inv);
        const int qw = quant8(mm.w ? 0.f : v.w, inv);
        qr[d4] = (unsigned)(uint8_t)qx | ((unsigned)(uint8_t)qy << 8) |
                 ((unsigned)(uint8_t)qz << 16) | ((unsigned)(uint8_t)qw << 24);
    }
}

__global__ __launch_bounds__(256)
void quant_w_kernel(const float* __restrict__ W, int8_t* __restrict__ wq,
                    float* __restrict__ sw) {
    __shared__ float red[256];
    const int o = blockIdx.x, tid = threadIdx.x;
    const float4* wr4 = (const float4*)(W + (size_t)o * Dk);
    float amax = 0.f;
    for (int d4 = tid; d4 < Dk / 4; d4 += 256) {
        const float4 v = wr4[d4];
        amax = fmaxf(amax, fmaxf(fmaxf(fabsf(v.x), fabsf(v.y)),
                                 fmaxf(fabsf(v.z), fabsf(v.w))));
    }
    red[tid] = amax; __syncthreads();
    for (int s = 128; s > 0; s >>= 1) {
        if (tid < s) red[tid] = fmaxf(red[tid], red[tid + s]);
        __syncthreads();
    }
    const float scale = red[0] * (1.0f / 127.0f);
    if (tid == 0) sw[o] = scale;
    const float inv = 1.0f / fmaxf(scale, EPSQ);
    unsigned* qr = (unsigned*)(wq + (size_t)o * Dk);
    for (int d4 = tid; d4 < Dk / 4; d4 += 256) {
        const float4 v = wr4[d4];
        qr[d4] = (unsigned)(uint8_t)quant8(v.x, inv) |
                 ((unsigned)(uint8_t)quant8(v.y, inv) << 8) |
                 ((unsigned)(uint8_t)quant8(v.z, inv) << 16) |
                 ((unsigned)(uint8_t)quant8(v.w, inv) << 24);
    }
}

__global__ __launch_bounds__(256)
void zero_tail_kernel(float* __restrict__ out) {
    float4* p = (float4*)(out + (size_t)blockIdx.x * OUTW + Ox);
    const float4 z = make_float4(0.f, 0.f, 0.f, 0.f);
    for (int i = threadIdx.x; i < (OUTW - Ox) / 4; i += 256) p[i] = z;
}

static constexpr int LDS_PITCH = 80;
static constexpr int TILE_BYTES = 128 * LDS_PITCH;
static constexpr int BUF_BYTES  = 2 * TILE_BYTES;

__global__ __launch_bounds__(256)
void gemm_i8_kernel(const uint8_t* __restrict__ xq, const uint8_t* __restrict__ wq,
                    const float* __restrict__ sx, const float* __restrict__ sw,
                    const float* __restrict__ x, const float* __restrict__ W,
                    const float* __restrict__ bias,
                    const unsigned* __restrict__ cnt, const unsigned* __restrict__ idx,
                    float* __restrict__ out) {
    __shared__ __align__(16) uint8_t smem[2 * BUF_BYTES];
    __shared__ int s_idx[64];
    __shared__ int s_n;
    if (threadIdx.x == 0) { int c = (int)*cnt; s_n = c > 64 ? 64 : c; }
    if (threadIdx.x < 64) s_idx[threadIdx.x] = (int)idx[threadIdx.x];

    const int tid  = threadIdx.x;
    const int lane = tid & 31;
    const int wave = tid >> 5;
    const int hi   = lane >> 4;
    const int ln   = lane & 15;
    const int wm   = wave & 3;
    const int wn   = wave >> 2;
    const int mb0  = blockIdx.y * 128;
    const int nb0  = blockIdx.x * 128;

    const unsigned smemBase = (unsigned)(uintptr_t)&smem[0];

    auto stage = [&](int kb, int buf) {
#pragma unroll
        for (int i = 0; i < 2; ++i) {
            const int u   = tid + i * 256;
            const int row = u >> 2;
            const int seg = (u & 3) << 4;
            const unsigned la = smemBase + (unsigned)(buf * BUF_BYTES + row * LDS_PITCH + seg);
            const unsigned long long ga =
                (unsigned long long)(xq + (size_t)(mb0 + row) * Dk + kb + seg);
            asm volatile("global_load_async_to_lds_b128 %0, %1, off"
                         :: "v"(la), "v"(ga) : "memory");
            const unsigned lb = la + (unsigned)TILE_BYTES;
            const unsigned long long gb =
                (unsigned long long)(wq + (size_t)(nb0 + row) * Dk + kb + seg);
            asm volatile("global_load_async_to_lds_b128 %0, %1, off"
                         :: "v"(lb), "v"(gb) : "memory");
            if (i == 0 && kb + 64 < Dk) {
                __builtin_prefetch((const void*)(ga + 64), 0, 0);
                __builtin_prefetch((const void*)(gb + 64), 0, 0);
            }
        }
    };

    v8i acc[2][4] = {};

    auto compute = [&](int buf) {
        const uint8_t* bufA = smem + buf * BUF_BYTES;
        const uint8_t* bufB = bufA + TILE_BYTES;

        v8i af[2];
#pragma unroll
        for (int ti = 0; ti < 2; ++ti) {
            const uint8_t* pa = bufA + (wm * 32 + ti * 16 + ln) * LDS_PITCH + hi * 16;
            const uint2 a0 = *(const uint2*)(pa +  0);
            const uint2 a1 = *(const uint2*)(pa +  8);
            const uint2 a2 = *(const uint2*)(pa + 32);
            const uint2 a3 = *(const uint2*)(pa + 40);
            af[ti][0] = (int)a0.x; af[ti][1] = (int)a0.y;
            af[ti][2] = (int)a1.x; af[ti][3] = (int)a1.y;
            af[ti][4] = (int)a2.x; af[ti][5] = (int)a2.y;
            af[ti][6] = (int)a3.x; af[ti][7] = (int)a3.y;
        }
        v8i bf[4];
#pragma unroll
        for (int tj = 0; tj < 4; ++tj) {
            const uint8_t* pb = bufB + (wn * 64 + tj * 16 + ln) * LDS_PITCH + hi * 16;
            const uint4 b0 = *(const uint4*)(pb +  0);
            const uint4 b1 = *(const uint4*)(pb + 32);
            bf[tj][0] = (int)b0.x; bf[tj][1] = (int)b0.y;
            bf[tj][2] = (int)b0.z; bf[tj][3] = (int)b0.w;
            bf[tj][4] = (int)b1.x; bf[tj][5] = (int)b1.y;
            bf[tj][6] = (int)b1.z; bf[tj][7] = (int)b1.w;
        }
#pragma unroll
        for (int ti = 0; ti < 2; ++ti)
#pragma unroll
            for (int tj = 0; tj < 4; ++tj)
                acc[ti][tj] = __builtin_amdgcn_wmma_i32_16x16x64_iu8(
                     true, af[ti],  true, bf[tj],
                    acc[ti][tj],  false,  false);
    };

    stage(0, 0);

    for (int kb = 0; kb < Dk - 64; kb += 64) {
        const int buf = (kb >> 6) & 1;
        stage(kb + 64, buf ^ 1);
        asm volatile("s_wait_asynccnt 0x4" ::: "memory");
        __syncthreads();
        compute(buf);
        __syncthreads();
    }
    asm volatile("s_wait_asynccnt 0x0" ::: "memory");
    __syncthreads();
    compute(((Dk - 64) >> 6) & 1);

#pragma unroll
    for (int ti = 0; ti < 2; ++ti) {
#pragma unroll
        for (int tj = 0; tj < 4; ++tj) {
            const int n = nb0 + wn * 64 + tj * 16 + ln;
            const float swn = sw[n];
            const float bn  = bias[n];
            const float* Wr = W + (size_t)n * Dk;
#pragma unroll
            for (int v = 0; v < 8; ++v) {
                const int m = mb0 + wm * 32 + ti * 16 + hi * 8 + v;
                float val = (float)acc[ti][tj][v] * sx[m] * swn;
                const float* xr = x + (size_t)m * Dk;
                for (int j = 0; j < s_n; ++j) {
                    const int d = s_idx[j];
                    val = fmaf(xr[d], Wr[d], val);
                }
                out[(size_t)m * OUTW + (size_t)n] = val + bn;
            }
        }
    }
}


extern "C" void kernel_launch(void* const* d_in, const int* in_sizes, int n_in,
                              void* d_out, int out_size, void* d_ws, size_t ws_size,
                              hipStream_t stream) {
    (void)in_sizes; (void)n_in; (void)out_size; (void)ws_size;
    const float* x = (const float*)d_in[0];
    const float* W = (const float*)d_in[1];
    const float* b = (const float*)d_in[2];
    float* out = (float*)d_out;
    uint8_t* ws = (uint8_t*)d_ws;

    unsigned* mask = (unsigned*)(ws + WS_MASK);
    unsigned* cnt  = (unsigned*)(ws + WS_CNT);
    unsigned* idx  = (unsigned*)(ws + WS_IDX);
    float*    sx   = (float*)(ws + WS_SX);
    float*    sw   = (float*)(ws + WS_SW);
    int8_t*   xq   = (int8_t*)(ws + WS_XQ);
    int8_t*   wq   = (int8_t*)(ws + WS_WQ);

    init_kernel<<<(Dk + 255) / 256, 256, 0, stream>>>(mask, cnt);
    detect_kernel<<<4096, 256, 0, stream>>>((const float4*)x, mask);
    collect_kernel<<<1, 256, 0, stream>>>(mask, cnt, idx);
    quant_x_kernel<<<Mx, 256, 0, stream>>>(x, mask, xq, sx);
    quant_w_kernel<<<Ox, 256, 0, stream>>>(W, wq, sw);
    zero_tail_kernel<<<Mx, 256, 0, stream>>>(out);

    dim3 grid(Ox / 128, Mx / 128);
    gemm_i8_kernel<<<grid, 256, 0, stream>>>((const uint8_t*)xq, (const uint8_t*)wq,
                                             sx, sw, x, W, b, cnt, idx, out);
}
